// MultiHeadAttention_17145509446464
// MI455X (gfx1250) — matrix operand order repaired and checked by reading, not compiled, not run on hardware
//
#include <hip/hip_runtime.h>
#include <hip/hip_bf16.h>
#include <math.h>

#define NB 16
#define NT 1024
#define NC 384
#define NH 6
#define ND 64

typedef __attribute__((ext_vector_type(16))) __bf16 v16bf;
typedef __attribute__((ext_vector_type(8)))  float  v8f;
typedef __attribute__((ext_vector_type(4)))  int    v4i;

union FragB { v16bf v; uint4 u[2]; };

#if __has_builtin(__builtin_amdgcn_global_load_async_to_lds_b128) && \
    __has_builtin(__builtin_amdgcn_s_wait_asynccnt)
#define HAS_ASYNC 1
#else
#define HAS_ASYNC 0
#endif

__device__ __forceinline__ unsigned short f2bf(float f) {
  unsigned int u = __builtin_bit_cast(unsigned int, f);
  u += 0x7FFFu + ((u >> 16) & 1u);           // round-to-nearest-even
  return (unsigned short)(u >> 16);
}

__device__ __forceinline__ FragB load_frag(const unsigned short* p0,
                                           const unsigned short* p1) {
  FragB f;
  f.u[0] = *reinterpret_cast<const uint4*>(p0);
  f.u[1] = *reinterpret_cast<const uint4*>(p1);
  return f;
}

// 16-byte global->LDS copy: async on CDNA5 toolchains that expose the builtin
__device__ __forceinline__ void cp16(unsigned short* lds, const unsigned short* g) {
#if HAS_ASYNC
  __builtin_amdgcn_global_load_async_to_lds_b128(
      (__attribute__((address_space(1))) v4i*)g,
      (__attribute__((address_space(3))) v4i*)lds, 0, 0);
#else
  *reinterpret_cast<uint4*>(lds) = *reinterpret_cast<const uint4*>(g);
#endif
}

__device__ __forceinline__ void wait_async4() {
#if HAS_ASYNC
  __builtin_amdgcn_s_wait_asynccnt(4);
#endif
}
__device__ __forceinline__ void wait_async0() {
#if HAS_ASYNC
  __builtin_amdgcn_s_wait_asynccnt(0);
#endif
}

#define WMMA_BF16(A, Bv, Cacc) \
  __builtin_amdgcn_wmma_f32_16x16x32_bf16(false, (A), false, (Bv), (short)0, (Cacc), false, false)

// ---------------- elementwise fp32 -> bf16 ----------------
__global__ void k_f32_to_bf16(const float* __restrict__ src,
                              unsigned short* __restrict__ dst, int n) {
  int i = blockIdx.x * blockDim.x + threadIdx.x;
  if (i < n) dst[i] = f2bf(src[i]);
}

// transpose [h][c][d] -> [h][d][c] (bf16) for wq/wk/wv so B-frags are k-contiguous
__global__ void k_w_transpose(const float* __restrict__ wq, const float* __restrict__ wk,
                              const float* __restrict__ wv,
                              unsigned short* __restrict__ wqT,
                              unsigned short* __restrict__ wkT,
                              unsigned short* __restrict__ wvT) {
  const int per = NH * NC * ND;
  int i = blockIdx.x * blockDim.x + threadIdx.x;
  if (i >= 3 * per) return;
  int which = i / per, r = i % per;
  int h = r / (ND * NC);
  int r2 = r % (ND * NC);
  int d = r2 / NC, c = r2 % NC;
  const float* s = which == 0 ? wq : (which == 1 ? wk : wv);
  unsigned short* o = which == 0 ? wqT : (which == 1 ? wkT : wvT);
  o[r] = f2bf(s[h * NC * ND + c * ND + d]);
}

// ---------------- QKV projection: one wave = 16 tokens x 64 dims ----------------
__global__ void k_qkv(const unsigned short* __restrict__ xb,
                      const unsigned short* __restrict__ wqT,
                      const unsigned short* __restrict__ wkT,
                      const unsigned short* __restrict__ wvT,
                      unsigned short* __restrict__ q,
                      unsigned short* __restrict__ kmat,
                      unsigned short* __restrict__ vT) {
  const int tile = blockIdx.x;
  const int b = tile >> 6;               // 64 row tiles per batch
  const int t0 = (tile & 63) << 4;
  const int h = blockIdx.y;
  const int which = blockIdx.z;
  const unsigned short* wT =
      (which == 0 ? wqT : (which == 1 ? wkT : wvT)) + (size_t)h * ND * NC;
  const int lane = threadIdx.x, ln = lane & 15, hh = lane >> 4;
  const unsigned short* xrow = xb + (size_t)(b * NT + t0 + ln) * NC;
  v8f acc[4] = {};
  for (int kk = 0; kk < NC; kk += 32) {
    FragB A = load_frag(xrow + kk + hh * 8, xrow + kk + 16 + hh * 8);
#pragma unroll
    for (int nt = 0; nt < 4; ++nt) {
      const unsigned short* bp = wT + (size_t)(nt * 16 + ln) * NC + kk + hh * 8;
      FragB Bf = load_frag(bp, bp + 16);
      acc[nt] = WMMA_BF16(A.v, Bf.v, acc[nt]);
    }
  }
  if (which == 2) {                       // V stored transposed [b][h][d][t]
    unsigned short* dst = vT + (size_t)(b * NH + h) * ND * NT;
#pragma unroll
    for (int nt = 0; nt < 4; ++nt)
#pragma unroll
      for (int r = 0; r < 8; ++r)
        dst[(size_t)(nt * 16 + ln) * NT + t0 + r + 8 * hh] = f2bf(acc[nt][r]);
  } else {                                // Q/K natural [b][h][t][d]
    unsigned short* dst = (which == 0 ? q : kmat) + ((size_t)(b * NH + h) * NT + t0) * ND;
#pragma unroll
    for (int nt = 0; nt < 4; ++nt)
#pragma unroll
      for (int r = 0; r < 8; ++r)
        dst[(size_t)(r + 8 * hh) * ND + nt * 16 + ln] = f2bf(acc[nt][r]);
  }
}

// ---------------- Flash attention ----------------
// 4 waves / block, 64 queries / block. K/V chunks (32 keys) staged in LDS with
// double-buffered async copies; each wave owns one 16-query row tile.
__global__ void __launch_bounds__(128) k_attn(const unsigned short* __restrict__ q,
                                              const unsigned short* __restrict__ kmat,
                                              const unsigned short* __restrict__ vT,
                                              unsigned short* __restrict__ att) {
  const int tid = threadIdx.x;
  const int w = tid >> 5;
  const int lane = tid & 31, ln = lane & 15, hh = lane >> 4;
  const int t0 = (blockIdx.x << 6) + (w << 4);
  const int h = blockIdx.y;
  const int b = blockIdx.z;
  const size_t bh = (size_t)(b * NH + h);
  const unsigned short* qp = q + bh * NT * ND;
  const unsigned short* kp = kmat + bh * NT * ND;
  const unsigned short* vp = vT + bh * ND * NT;

  __shared__ unsigned short Kt[2][32 * 64];   // [s][d] chunk, 4KB each
  __shared__ unsigned short Vt[2][64 * 32];   // [d][s] chunk, 4KB each
  __shared__ unsigned short Pl[4][16 * 32];   // per-wave P transpose buffer
  unsigned short* Pw = Pl[w];

  FragB QA[2];
  {
    const unsigned short* qrow = qp + (size_t)(t0 + ln) * ND;
#pragma unroll
    for (int kd = 0; kd < 2; ++kd)
      QA[kd] = load_frag(qrow + kd * 32 + hh * 8, qrow + kd * 32 + 16 + hh * 8);
  }
  v8f acc[4] = {};
  float m_i[8], l_i[8];
#pragma unroll
  for (int r = 0; r < 8; ++r) { m_i[r] = -INFINITY; l_i[r] = 0.f; }

  const float sc = 0.05103103630798287f;   // 1/sqrt(384) (ref scales by embed dim)

  // cooperative chunk loader: 4 x b128 per thread -> 4 async ops per wave
  auto issue = [&](int s0, int buf) {
#pragma unroll
    for (int it = 0; it < 2; ++it) {
      int idx = it * 128 + tid;                         // 256 x 16B = K chunk
      cp16(&Kt[buf][idx * 8], kp + (size_t)s0 * ND + idx * 8);
    }
#pragma unroll
    for (int it = 0; it < 2; ++it) {
      int idx = it * 128 + tid;                         // 256 x 16B = V chunk
      int d = idx >> 2, seg = idx & 3;
      cp16(&Vt[buf][d * 32 + seg * 8], vp + (size_t)d * NT + s0 + seg * 8);
    }
  };

  issue(0, 0);
  for (int c = 0; c < NT / 32; ++c) {
    const int buf = c & 1;
    if (c + 1 < NT / 32) {
      issue((c + 1) * 32, buf ^ 1);
      wait_async4();                 // chunk c's 4 ops drained, next 4 in flight
    } else {
      wait_async0();
    }
    __syncthreads();                 // all waves' portions of Kt/Vt[buf] visible

    v8f S0 = {}, S1 = {};
#pragma unroll
    for (int kd = 0; kd < 2; ++kd) {
      const unsigned short* kr0 = &Kt[buf][ln * 64 + kd * 32 + hh * 8];
      FragB B0 = load_frag(kr0, kr0 + 16);
      S0 = WMMA_BF16(QA[kd].v, B0.v, S0);
      const unsigned short* kr1 = &Kt[buf][(16 + ln) * 64 + kd * 32 + hh * 8];
      FragB B1 = load_frag(kr1, kr1 + 16);
      S1 = WMMA_BF16(QA[kd].v, B1.v, S1);
    }
#pragma unroll
    for (int r = 0; r < 8; ++r) {
      float x0 = S0[r] * sc, x1 = S1[r] * sc;
      float mx = fmaxf(x0, x1);               // row lives across 16 lanes of a half
      mx = fmaxf(mx, __shfl_xor(mx, 1, 32));
      mx = fmaxf(mx, __shfl_xor(mx, 2, 32));
      mx = fmaxf(mx, __shfl_xor(mx, 4, 32));
      mx = fmaxf(mx, __shfl_xor(mx, 8, 32));
      float mnew = fmaxf(m_i[r], mx);
      float resc = __expf(m_i[r] - mnew);
      float p0 = __expf(x0 - mnew);
      float p1 = __expf(x1 - mnew);
      float rs = p0 + p1;
      rs += __shfl_xor(rs, 1, 32);
      rs += __shfl_xor(rs, 2, 32);
      rs += __shfl_xor(rs, 4, 32);
      rs += __shfl_xor(rs, 8, 32);
      l_i[r] = l_i[r] * resc + rs;
      m_i[r] = mnew;
#pragma unroll
      for (int nt = 0; nt < 4; ++nt) acc[nt][r] *= resc;
      int mrow = r + 8 * hh;                  // transpose P (C layout -> A layout)
      Pw[mrow * 32 + ln]      = f2bf(p0);
      Pw[mrow * 32 + 16 + ln] = f2bf(p1);
    }
    __syncthreads();
    FragB PA;
    {
      const unsigned short* pr = &Pw[ln * 32];
      PA.u[0] = *reinterpret_cast<const uint4*>(pr + hh * 8);
      PA.u[1] = *reinterpret_cast<const uint4*>(pr + 16 + hh * 8);
    }
#pragma unroll
    for (int nt = 0; nt < 4; ++nt) {
      const unsigned short* vr = &Vt[buf][(nt * 16 + ln) * 32 + hh * 8];
      FragB VB = load_frag(vr, vr + 16);
      acc[nt] = WMMA_BF16(PA.v, VB.v, acc[nt]);
    }
    __syncthreads();                 // reads of buf done before it is refilled
  }
#pragma unroll
  for (int r = 0; r < 8; ++r) {
    float inv = 1.0f / l_i[r];
    size_t rowbase = (size_t)(b * NT + t0 + r + 8 * hh) * NC + h * ND;
#pragma unroll
    for (int nt = 0; nt < 4; ++nt)
      att[rowbase + nt * 16 + ln] = f2bf(acc[nt][r] * inv);
  }
}

// ---------------- output projection: out = att @ wp^T + bp (fp32 out) ----------------
__global__ void k_proj(const unsigned short* __restrict__ att,
                       const unsigned short* __restrict__ wpb,
                       const float* __restrict__ bp,
                       float* __restrict__ out) {
  const int row0 = blockIdx.x << 4;
  const int e0 = blockIdx.y << 6;
  const int lane = threadIdx.x, ln = lane & 15, hh = lane >> 4;
  const unsigned short* arow = att + (size_t)(row0 + ln) * NC;
  v8f acc[4] = {};
  for (int kk = 0; kk < NC; kk += 32) {
    FragB A = load_frag(arow + kk + hh * 8, arow + kk + 16 + hh * 8);
#pragma unroll
    for (int nt = 0; nt < 4; ++nt) {
      const unsigned short* bpp = wpb + (size_t)(e0 + nt * 16 + ln) * NC + kk + hh * 8;
      FragB Bf = load_frag(bpp, bpp + 16);
      acc[nt] = WMMA_BF16(A.v, Bf.v, acc[nt]);
    }
  }
#pragma unroll
  for (int nt = 0; nt < 4; ++nt) {
    int e = e0 + nt * 16 + ln;
    float bv = bp[e];
#pragma unroll
    for (int r = 0; r < 8; ++r)
      out[(size_t)(row0 + r + 8 * hh) * NC + e] = acc[nt][r] + bv;
  }
}

extern "C" void kernel_launch(void* const* d_in, const int* in_sizes, int n_in,
                              void* d_out, int out_size, void* d_ws, size_t ws_size,
                              hipStream_t stream) {
  const float* x  = (const float*)d_in[0];
  const float* wq = (const float*)d_in[1];
  const float* wk = (const float*)d_in[2];
  const float* wv = (const float*)d_in[3];
  const float* wp = (const float*)d_in[4];
  const float* bp = (const float*)d_in[5];
  float* out = (float*)d_out;

  char* w = (char*)d_ws;
  unsigned short* xb  = (unsigned short*)w; w += (size_t)NB * NT * NC * 2;
  unsigned short* q   = (unsigned short*)w; w += (size_t)NB * NH * NT * ND * 2;
  unsigned short* km  = (unsigned short*)w; w += (size_t)NB * NH * NT * ND * 2;
  unsigned short* vT  = (unsigned short*)w; w += (size_t)NB * NH * ND * NT * 2;
  unsigned short* att = (unsigned short*)w; w += (size_t)NB * NT * NC * 2;
  unsigned short* wqT = (unsigned short*)w; w += (size_t)NH * ND * NC * 2;
  unsigned short* wkT = (unsigned short*)w; w += (size_t)NH * ND * NC * 2;
  unsigned short* wvT = (unsigned short*)w; w += (size_t)NH * ND * NC * 2;
  unsigned short* wpb = (unsigned short*)w; w += (size_t)NC * NC * 2;

  {
    int n = NB * NT * NC;
    k_f32_to_bf16<<<(n + 255) / 256, 256, 0, stream>>>(x, xb, n);
  }
  {
    int n = NC * NC;
    k_f32_to_bf16<<<(n + 255) / 256, 256, 0, stream>>>(wp, wpb, n);
  }
  {
    int n = 3 * NH * NC * ND;
    k_w_transpose<<<(n + 255) / 256, 256, 0, stream>>>(wq, wk, wv, wqT, wkT, wvT);
  }
  k_qkv <<<dim3(NB * NT / 16, NH, 3), 32, 0, stream>>>(xb, wqT, wkT, wvT, q, km, vT);
  k_attn<<<dim3(NT / 64, NH, NB),    128, 0, stream>>>(q, km, vT, att);
  k_proj<<<dim3(NB * NT / 16, NC / 64), 32, 0, stream>>>(att, wpb, bp, out);
}
